// CausalSelfAttention_48627619725750
// MI455X (gfx1250) — matrix operand order repaired and checked by reading, not compiled, not run on hardware
//
#include <hip/hip_runtime.h>
#include <hip/hip_bf16.h>

typedef __attribute__((ext_vector_type(8)))  float   v8f;
typedef __attribute__((ext_vector_type(8)))  __bf16  v8bf;
typedef __attribute__((ext_vector_type(16))) __bf16  v16bf;

static __device__ __forceinline__ v16bf join16(v8bf lo, v8bf hi) {
  return __builtin_shufflevector(lo, hi, 0,1,2,3,4,5,6,7,8,9,10,11,12,13,14,15);
}

static __device__ __forceinline__ v8f wmma_bf16(v16bf a, v16bf b, v8f c) {
  return __builtin_amdgcn_wmma_f32_16x16x32_bf16(false, a, false, b, (short)0, c, false, false);
}

// ---------------------------------------------------------------------------
// GEMM with fused bias: Out[M,N] = A[M,K] * B[K,N] + bias[N]
// A: fp32 or bf16 (template), B: fp32 (converted to bf16 in staging),
// Out: bf16 or fp32 (template). Block tile 128x128, K-step 32, 8 waves.
// Register-prefetch software pipeline: next K-panel's global loads are issued
// before the WMMA block so load latency overlaps the matrix math.
// ---------------------------------------------------------------------------
template <bool A_BF16, bool OUT_BF16>
__global__ __launch_bounds__(256) void gemm_bias_kernel(
    const void* __restrict__ Ap, const float* __restrict__ Bp,
    const float* __restrict__ bias, void* __restrict__ Outp,
    int M, int N, int K)
{
  __shared__ __bf16 lA[128 * 32];   // [m][k]
  __shared__ __bf16 lB[128 * 32];   // [n][k]  (B staged transposed)

  const int tid  = threadIdx.x;
  const int lane = tid & 31;
  const int wave = __builtin_amdgcn_readfirstlane(tid >> 5);  // SGPR wave id
  const int hi   = lane >> 4;          // half-wave select
  const int ll   = lane & 15;
  const int wm   = (wave >> 2) * 64;   // wave M offset in tile (scalar)
  const int wn   = (wave & 3) * 32;    // wave N offset in tile (scalar)
  const int mBase = blockIdx.y * 128;
  const int nBase = blockIdx.x * 128;

  const float*  Af = (const float*)Ap;
  const __bf16* Ab = (const __bf16*)Ap;

  // Per-thread staging coordinates (fixed across K panels)
  const int sAm = tid >> 5, sAk = tid & 31;      // A: 8 rows apart per i
  const int sBk = tid >> 7, sBn = tid & 127;     // B: 2 k-rows apart per i

  v8f acc[4][2];
  for (int i = 0; i < 4; ++i)
    for (int j = 0; j < 2; ++j)
      acc[i][j] = (v8f)0.0f;

  float ra[16], rb[16];

  // Prologue: fetch panel 0 into registers, deposit to LDS.
  for (int i = 0; i < 16; ++i) {
    size_t g = (size_t)(mBase + sAm + i * 8) * K + sAk;
    ra[i] = A_BF16 ? (float)Ab[g] : Af[g];
    rb[i] = Bp[(size_t)(sBk + i * 2) * N + (nBase + sBn)];
  }
  for (int i = 0; i < 16; ++i) {
    lA[(sAm + i * 8) * 32 + sAk] = (__bf16)ra[i];
    lB[sBn * 32 + (sBk + i * 2)] = (__bf16)rb[i];
  }
  __syncthreads();

  const int KT = K / 32;
  for (int kt = 0; kt < KT; ++kt) {
    // Issue next panel's global loads (latency hidden behind WMMAs below)
    if (kt + 1 < KT) {
      int k0 = (kt + 1) * 32;
      for (int i = 0; i < 16; ++i) {
        size_t g = (size_t)(mBase + sAm + i * 8) * K + (k0 + sAk);
        ra[i] = A_BF16 ? (float)Ab[g] : Af[g];
        rb[i] = Bp[(size_t)(k0 + sBk + i * 2) * N + (nBase + sBn)];
      }
    }

    // Compute on current LDS panel
    v16bf afr[4], bfr[2];
    for (int mf = 0; mf < 4; ++mf) {
      int off = (wm + mf * 16 + ll) * 32 + hi * 8;
      afr[mf] = join16(*(const v8bf*)&lA[off], *(const v8bf*)&lA[off + 16]);
    }
    for (int nf = 0; nf < 2; ++nf) {
      int off = (wn + nf * 16 + ll) * 32 + hi * 8;
      bfr[nf] = join16(*(const v8bf*)&lB[off], *(const v8bf*)&lB[off + 16]);
    }
    for (int mf = 0; mf < 4; ++mf)
      for (int nf = 0; nf < 2; ++nf)
        acc[mf][nf] = wmma_bf16(afr[mf], bfr[nf], acc[mf][nf]);

    __syncthreads();   // all reads of current panel retired

    if (kt + 1 < KT) {
      for (int i = 0; i < 16; ++i) {
        lA[(sAm + i * 8) * 32 + sAk] = (__bf16)ra[i];
        lB[sBn * 32 + (sBk + i * 2)] = (__bf16)rb[i];
      }
      __syncthreads();
    }
  }

  // Epilogue: C/D layout row = v + 8*hi, col = lane&15
  for (int nf = 0; nf < 2; ++nf) {
    int col = nBase + wn + nf * 16 + ll;
    float bv = bias[col];
    for (int mf = 0; mf < 4; ++mf) {
      for (int v = 0; v < 8; ++v) {
        int row = mBase + wm + mf * 16 + v + 8 * hi;
        float o = acc[mf][nf][v] + bv;
        if (OUT_BF16) ((__bf16*)Outp)[(size_t)row * N + col] = (__bf16)o;
        else          ((float*)Outp)[(size_t)row * N + col]  = o;
      }
    }
  }
}

// ---------------------------------------------------------------------------
// Flash attention, transposed formulation:
//   S^T = K * Q^T   (keys in rows -> softmax stats are per-lane scalars)
//   O^T = V^T * P^T (accumulated in f32)
// One wave owns 16 queries; block = 8 waves = 128 queries of one (b,h).
// ---------------------------------------------------------------------------
__global__ __launch_bounds__(256) void attn_kernel(const __bf16* __restrict__ qkv,
                                                   __bf16* __restrict__ y)
{
  constexpr int T = 2048, C = 1024, C3 = 3072;
  __shared__ __bf16 lK [128 * 64];   // [key][d]
  __shared__ __bf16 lVT[64 * 128];   // [d][key]

  const int tid  = threadIdx.x;
  const int lane = tid & 31;
  const int wave = __builtin_amdgcn_readfirstlane(tid >> 5);  // SGPR wave id
  const int hi   = lane >> 4;
  const int ll   = lane & 15;

  const int qb = blockIdx.x & 15;          // query block (T/128 = 16)
  const int h  = (blockIdx.x >> 4) & 15;   // head
  const int b  = blockIdx.x >> 8;          // batch

  const int qBase = qb * 128 + wave * 16;  // scalar
  const int q = qBase + ll;

  // Q^T B-fragments: lane col = query, d = 8*hi + {0..7} and 16 + 8*hi + {0..7}
  v16bf qf[2];
  for (int s = 0; s < 2; ++s) {
    const __bf16* p = qkv + ((size_t)(b * T + q)) * C3 + h * 64 + s * 32 + hi * 8;
    qf[s] = join16(*(const v8bf*)p, *(const v8bf*)(p + 16));
  }

  v8f o[4];
  for (int i = 0; i < 4; ++i) o[i] = (v8f)0.0f;
  float m = -1e30f, l = 0.0f;
  const float scale = 0.125f;   // 1/sqrt(64)

  for (int kb = 0; kb <= qb; ++kb) {
    // Stage K [key][d] and V transposed [d][key]
    for (int i = 0; i < 4; ++i) {
      int cidx = tid + i * 256;
      int key = cidx >> 3;
      int dc  = (cidx & 7) * 8;
      const __bf16* kp = qkv + ((size_t)(b * T + kb * 128 + key)) * C3 + C + h * 64 + dc;
      *(v8bf*)&lK[key * 64 + dc] = *(const v8bf*)kp;
      v8bf vv = *(const v8bf*)(kp + C);
      for (int j = 0; j < 8; ++j) lVT[(dc + j) * 128 + key] = vv[j];
    }
    __syncthreads();

    for (int c = 0; c < 4; ++c) {
      const int kcb = kb * 128 + c * 32;   // scalar
      if (kcb > qBase + 15) break;         // scalar branch: chunk above diagonal

      // S^T accumulators for key subtiles [kcb..+15] and [kcb+16..+31]
      v8f s0 = (v8f)0.0f, s1 = (v8f)0.0f;
      for (int s = 0; s < 2; ++s) {
        int off0 = (c * 32 + ll) * 64 + s * 32 + hi * 8;
        v16bf a0 = join16(*(const v8bf*)&lK[off0], *(const v8bf*)&lK[off0 + 16]);
        int off1 = off0 + 16 * 64;
        v16bf a1 = join16(*(const v8bf*)&lK[off1], *(const v8bf*)&lK[off1 + 16]);
        s0 = wmma_bf16(a0, qf[s], s0);
        s1 = wmma_bf16(a1, qf[s], s1);
      }

      // Scale + causal mask (S^T row = key = kcb + v + 8*hi, col = query = q)
      float mt = -1e30f;
      for (int v = 0; v < 8; ++v) {
        int k0 = kcb + v + 8 * hi;
        float x0 = (k0      <= q) ? s0[v] * scale : -1e30f;
        float x1 = (k0 + 16 <= q) ? s1[v] * scale : -1e30f;
        s0[v] = x0; s1[v] = x1;
        mt = fmaxf(mt, fmaxf(x0, x1));
      }
      mt = fmaxf(mt, __shfl_xor(mt, 16, 32));
      float mn = fmaxf(m, mt);
      float f = __expf(m - mn);
      m = mn;
      l *= f;
      for (int dg = 0; dg < 4; ++dg)
        for (int v = 0; v < 8; ++v) o[dg][v] *= f;

      float rs = 0.0f;
      for (int v = 0; v < 8; ++v) {
        s0[v] = __expf(s0[v] - mn);
        s1[v] = __expf(s1[v] - mn);
        rs += s0[v] + s1[v];
      }
      l += rs + __shfl_xor(rs, 16, 32);

      // Pack P^T (32 keys x 16 queries) into a B-fragment:
      // lanes hi=0 need keys 0..15 (own s0 + partner s0), hi=1 keys 16..31.
      v16bf pb;
      for (int v = 0; v < 8; ++v) {
        float send = hi ? s0[v] : s1[v];
        float recv = __shfl_xor(send, 16, 32);
        float lo  = hi ? recv  : s0[v];
        float hiv = hi ? s1[v] : recv;
        pb[v]     = (__bf16)lo;
        pb[v + 8] = (__bf16)hiv;
      }

      // O^T += V^T(16d x 32k) * P^T(32k x 16q), 4 d-groups
      for (int dg = 0; dg < 4; ++dg) {
        int off = (dg * 16 + ll) * 128 + c * 32 + hi * 16;
        v16bf vf = join16(*(const v8bf*)&lVT[off], *(const v8bf*)&lVT[off + 8]);
        o[dg] = wmma_bf16(vf, pb, o[dg]);
      }
    }
    __syncthreads();
  }

  // Write y[b*T+q][h*64+d] = O^T / l  (O^T layout: d = dg*16 + v + 8*hi)
  float inv = 1.0f / l;
  for (int dg = 0; dg < 4; ++dg)
    for (int v = 0; v < 8; ++v) {
      int d = dg * 16 + v + 8 * hi;
      y[((size_t)(b * T + q)) * C + h * 64 + d] = (__bf16)(o[dg][v] * inv);
    }
}

// ---------------------------------------------------------------------------
extern "C" void kernel_launch(void* const* d_in, const int* in_sizes, int n_in,
                              void* d_out, int out_size, void* d_ws, size_t ws_size,
                              hipStream_t stream) {
  const float* x      = (const float*)d_in[0];
  const float* w_attn = (const float*)d_in[1];
  const float* b_attn = (const float*)d_in[2];
  const float* w_proj = (const float*)d_in[3];
  const float* b_proj = (const float*)d_in[4];

  constexpr int B = 4, T = 2048, C = 1024;
  constexpr size_t QKV_ELEMS = (size_t)B * T * 3 * C;   // 25.2M * 2B = 48MB

  __bf16* qkv = (__bf16*)d_ws;
  __bf16* y   = (__bf16*)((char*)d_ws + QKV_ELEMS * sizeof(__bf16));

  // 1) qkv = x @ w_attn + b_attn   (fp32 in, bf16 out)
  gemm_bias_kernel<false, true><<<dim3(3 * C / 128, B * T / 128), 256, 0, stream>>>(
      x, w_attn, b_attn, qkv, B * T, 3 * C, C);

  // 2) flash attention per (b, h, 128-query block)
  attn_kernel<<<dim3(B * 16 * (T / 128)), 256, 0, stream>>>(qkv, y);

  // 3) out = y @ w_proj + b_proj   (bf16 in, fp32 out)
  gemm_bias_kernel<true, false><<<dim3(C / 128, B * T / 128), 256, 0, stream>>>(
      y, w_proj, b_proj, d_out, B * T, C, C);
}
